// UnimolDescriminator_38070590112238
// MI455X (gfx1250) — matrix operand order repaired and checked by reading, not compiled, not run on hardware
//
#include <hip/hip_runtime.h>
#include <hip/hip_bf16.h>
#include <math.h>

// ---- problem constants (match reference) ----
#define B_ 16
#define L_ 256
#define D_ 768
#define H_ 12
#define NL_ 6
#define DH_ 64
#define DF_ 3072
#define VN_ 128
#define VE_ 64
#define BL_ (B_*L_)
#define BHLL_ ((long long)B_*H_*L_*L_)
#define SCALE_F 0.125f   // 1/sqrt(64)

typedef __bf16 bf16_t;
typedef __attribute__((ext_vector_type(16))) __bf16 bf16x16;
typedef __attribute__((ext_vector_type(8)))  __bf16 bf16x8;
typedef __attribute__((ext_vector_type(8)))  float  f32x8;

__device__ __forceinline__ bf16_t f2bf(float f) {
    unsigned u = __builtin_bit_cast(unsigned, f);
    unsigned r = u + 0x7FFFu + ((u >> 16) & 1u);   // round-to-nearest-even
    unsigned short h = (unsigned short)(r >> 16);
    return __builtin_bit_cast(bf16_t, h);
}

// ---------------- elementwise / setup kernels ----------------

__global__ void k_f2bf(const float* __restrict__ in, bf16_t* __restrict__ out, long long n) {
    long long i = (long long)blockIdx.x * blockDim.x + threadIdx.x;
    if (i < n) out[i] = f2bf(in[i]);
}

__global__ void k_node_gather(const int* __restrict__ node, const float* __restrict__ table,
                              float* __restrict__ n0) {
    long long i = (long long)blockIdx.x * blockDim.x + threadIdx.x;
    if (i >= (long long)BL_ * D_) return;
    long long row = i / D_; int d = (int)(i % D_);
    n0[i] = table[(long long)node[row] * D_ + d];
}

// e0[b,h,l,m] = pad ? -inf : edge_table[edge[b,l,m]][h]
// e1[b,h,l,m] = pad ? -inf : sum_e edge_emb[b,l,m,e]*edge_table[e][h]
__global__ void k_edge_embed(const int* __restrict__ edge, const int* __restrict__ node,
                             const float* __restrict__ edge_emb, const float* __restrict__ edge_table,
                             float* __restrict__ e0, float* __restrict__ e1) {
    long long idx = (long long)blockIdx.x * blockDim.x + threadIdx.x;   // over B*L*L
    if (idx >= (long long)B_ * L_ * L_) return;
    int m = (int)(idx % L_);
    long long t = idx / L_;
    int l = (int)(t % L_);
    int b = (int)(t / L_);
    bool pad = (node[b * L_ + m] == 0);
    int id = edge[idx];
    float acc[H_];
#pragma unroll
    for (int h = 0; h < H_; ++h) acc[h] = 0.f;
    const float* emb = edge_emb + idx * VE_;
    for (int e = 0; e < VE_; ++e) {
        float v = emb[e];
#pragma unroll
        for (int h = 0; h < H_; ++h) acc[h] += v * edge_table[e * H_ + h];
    }
#pragma unroll
    for (int h = 0; h < H_; ++h) {
        long long o = (((long long)(b * H_ + h) * L_) + l) * L_ + m;
        e0[o] = pad ? -__builtin_inff() : edge_table[id * H_ + h];
        e1[o] = pad ? -__builtin_inff() : acc[h];
    }
}

__global__ void k_edge_update(const float* __restrict__ e, const float* __restrict__ t,
                              const float* __restrict__ t2, float* __restrict__ out, long long n) {
    long long i = (long long)blockIdx.x * blockDim.x + threadIdx.x;
    if (i < n) out[i] = e[i] + t[i] + t2[i];
}

// V transpose: qkv[B,L,3D] cols [2D..3D) -> vT[B,H,DH,L]  (bf16 -> bf16)
__global__ void k_transpose_v(const bf16_t* __restrict__ qkv, bf16_t* __restrict__ vT) {
    long long i = (long long)blockIdx.x * blockDim.x + threadIdx.x;   // over B*H*DH*L
    if (i >= (long long)BL_ * D_) return;
    int l = (int)(i % L_);
    long long t = i / L_;
    int d = (int)(t % DH_); t /= DH_;
    int h = (int)(t % H_);
    int b = (int)(t / H_);
    vT[i] = qkv[((long long)b * L_ + l) * (3 * D_) + 2 * D_ + h * DH_ + d];
}

// node_table [VN,D] f32 -> transposed bf16 [D,VN]
__global__ void k_transpose_ntab(const float* __restrict__ tab, bf16_t* __restrict__ out) {
    long long i = (long long)blockIdx.x * blockDim.x + threadIdx.x;   // over D*VN
    if (i >= (long long)D_ * VN_) return;
    int v = (int)(i % VN_);
    int d = (int)(i / VN_);
    out[i] = f2bf(tab[(long long)v * D_ + d]);
}

// LayerNorm over last dim (d), one block per row, bf16 output
__global__ void k_ln_bf16(const float* __restrict__ x, const float* __restrict__ g,
                          const float* __restrict__ b, bf16_t* __restrict__ out, int d) {
    int row = blockIdx.x;
    const float* xr = x + (long long)row * d;
    __shared__ float red[256];
    int tid = threadIdx.x;
    float s = 0.f;
    for (int i = tid; i < d; i += blockDim.x) s += xr[i];
    red[tid] = s; __syncthreads();
    for (int o = 128; o > 0; o >>= 1) { if (tid < o) red[tid] += red[tid + o]; __syncthreads(); }
    float mu = red[0] / (float)d; __syncthreads();
    float s2 = 0.f;
    for (int i = tid; i < d; i += blockDim.x) { float v = xr[i] - mu; s2 += v * v; }
    red[tid] = s2; __syncthreads();
    for (int o = 128; o > 0; o >>= 1) { if (tid < o) red[tid] += red[tid + o]; __syncthreads(); }
    float rinv = rsqrtf(red[0] / (float)d + 1e-5f); __syncthreads();
    bf16_t* orow = out + (long long)row * d;
    for (int i = tid; i < d; i += blockDim.x)
        orow[i] = f2bf((xr[i] - mu) * rinv * g[i] + b[i]);
}

// softmax over rows of length 256; one wave32 per row; bf16 probs out
__global__ void k_softmax_bf16(const float* __restrict__ t, const float* __restrict__ e,
                               bf16_t* __restrict__ p, long long nrows) {
    long long row = (long long)blockIdx.x * (blockDim.x >> 5) + (threadIdx.x >> 5);
    if (row >= nrows) return;
    int lane = threadIdx.x & 31;
    const float* tr = t + row * L_;
    const float* er = e + row * L_;
    float v[8];
    float m = -__builtin_inff();
#pragma unroll
    for (int j = 0; j < 8; ++j) { v[j] = tr[lane * 8 + j] + er[lane * 8 + j]; m = fmaxf(m, v[j]); }
    for (int o = 16; o > 0; o >>= 1) m = fmaxf(m, __shfl_xor(m, o, 32));
    float s = 0.f;
#pragma unroll
    for (int j = 0; j < 8; ++j) { v[j] = expf(v[j] - m); s += v[j]; }
    for (int o = 16; o > 0; o >>= 1) s += __shfl_xor(s, o, 32);
    float inv = 1.f / s;
    bf16_t* pr = p + row * L_;
#pragma unroll
    for (int j = 0; j < 8; ++j) pr[lane * 8 + j] = f2bf(v[j] * inv);
}

// masked mean over L then dot with W_p; one block per batch element
__global__ void k_pool_proj(const float* __restrict__ n1, const int* __restrict__ node,
                            const float* __restrict__ Wp, const float* __restrict__ bp,
                            float* __restrict__ out) {
    int b = blockIdx.x, tid = threadIdx.x;
    __shared__ float red[256];
    __shared__ float msum;
    if (tid == 0) {
        float c = 0.f;
        for (int l = 0; l < L_; ++l) c += (node[b * L_ + l] != 0) ? 1.f : 0.f;
        msum = c;
    }
    __syncthreads();
    float partial = 0.f;
    for (int d = tid; d < D_; d += 256) {
        float acc = 0.f;
        for (int l = 0; l < L_; ++l)
            if (node[b * L_ + l] != 0) acc += n1[((long long)b * L_ + l) * D_ + d];
        partial += acc * Wp[d];
    }
    red[tid] = partial; __syncthreads();
    for (int o = 128; o > 0; o >>= 1) { if (tid < o) red[tid] += red[tid + o]; __syncthreads(); }
    if (tid == 0) out[b] = red[0] / msum + bp[0];
}

// ---------------- generic batched WMMA GEMM, 32x64 macro-tile per wave ----------------
// out[row,col] = act( scale * sum_k A[row,k]*W[col,k] + bias[col] + resid[row,col] )
// B operand is ALWAYS W-layout: [N,K] row-major (contiguous along K).
// Each wave32: 2 (M) x 4 (N) = 8 accumulators of 16x16, K stepped by 32 via
// v_wmma_f32_16x16x32_bf16; A frags reused 4x, B frags reused 2x.
// __launch_bounds__(256, 1): few fat waves per SIMD -> no VGPR spills, WMMA
// latency hidden by the compiler's software pipelining of the b128 loads.
__global__ void __launch_bounds__(256, 1)
k_wmma_gemm(const bf16_t* __restrict__ A, const bf16_t* __restrict__ Bm,
            const float* __restrict__ bias, const float* __restrict__ resid,
            float* __restrict__ outF, bf16_t* __restrict__ outH,
            int M, int N, int K, int lda, int ldb, int ldr, int ldo, int ldoh,
            float scale, int actGelu, int hh,
            long long sAb, long long sAh, long long sBb, long long sBh,
            long long sOb, long long sOh, long long sOHb, long long sOHh) {
    int z = blockIdx.y;
    int zb = z / hh, zh = z % hh;
    A  += zb * sAb + zh * sAh;
    Bm += zb * sBb + zh * sBh;
    if (outF) outF += zb * sOb + zh * sOh;
    if (outH) outH += zb * sOHb + zh * sOHh;

    int nTn = N >> 6;                      // 64-wide N macro-tiles
    int tiles = (M >> 5) * nTn;            // 32-tall M macro-tiles
    int wave = blockIdx.x * (blockDim.x >> 5) + (threadIdx.x >> 5);
    if (wave >= tiles) return;             // wave-uniform: EXEC all-ones inside
    int tm = (wave / nTn) << 5;
    int tn = (wave % nTn) << 6;
    int lane = threadIdx.x & 31;
    int half = lane >> 4;
    int r = lane & 15;

    f32x8 acc[2][4];
#pragma unroll
    for (int mi = 0; mi < 2; ++mi)
#pragma unroll
        for (int ni = 0; ni < 4; ++ni) acc[mi][ni] = (f32x8){0.f,0.f,0.f,0.f,0.f,0.f,0.f,0.f};

    // A fragment bases: lanes 0-15 cover K k0+0..7 / k0+16..23; lanes 16-31 are +8.
    const bf16_t* aP0 = A + (long long)(tm + r) * lda + half * 8;
    const bf16_t* aP1 = aP0 + 16LL * lda;
    // B (W-layout) fragment bases: same K split as A (k0+0..7 / k0+16..23; lanes 16-31 +8).
    const bf16_t* bP0 = Bm + (long long)(tn + r) * ldb + half * 8;
    const bf16_t* bP1 = bP0 + 16LL * ldb;
    const bf16_t* bP2 = bP0 + 32LL * ldb;
    const bf16_t* bP3 = bP0 + 48LL * ldb;

    for (int k0 = 0; k0 < K; k0 += 32) {
        bf16x16 af[2], bf[4];
        {
            bf16x8 lo0 = *(const bf16x8*)(aP0 + k0);
            bf16x8 hi0 = *(const bf16x8*)(aP0 + k0 + 16);
            bf16x8 lo1 = *(const bf16x8*)(aP1 + k0);
            bf16x8 hi1 = *(const bf16x8*)(aP1 + k0 + 16);
#pragma unroll
            for (int i = 0; i < 8; ++i) {
                af[0][i] = lo0[i]; af[0][8 + i] = hi0[i];
                af[1][i] = lo1[i]; af[1][8 + i] = hi1[i];
            }
        }
        {
            const bf16_t* bp[4] = {bP0 + k0, bP1 + k0, bP2 + k0, bP3 + k0};
#pragma unroll
            for (int ni = 0; ni < 4; ++ni) {
                bf16x8 lo = *(const bf16x8*)(bp[ni]);
                bf16x8 hi = *(const bf16x8*)(bp[ni] + 16);
#pragma unroll
                for (int i = 0; i < 8; ++i) { bf[ni][i] = lo[i]; bf[ni][8 + i] = hi[i]; }
            }
        }
#pragma unroll
        for (int mi = 0; mi < 2; ++mi)
#pragma unroll
            for (int ni = 0; ni < 4; ++ni)
                acc[mi][ni] = __builtin_amdgcn_wmma_f32_16x16x32_bf16(
                    false, af[mi], false, bf[ni], (short)0, acc[mi][ni], false, false);
    }

#pragma unroll
    for (int mi = 0; mi < 2; ++mi) {
#pragma unroll
        for (int ni = 0; ni < 4; ++ni) {
            int col = tn + ni * 16 + r;
            float bcol = bias ? bias[col] : 0.f;
#pragma unroll
            for (int i = 0; i < 8; ++i) {
                int row = tm + mi * 16 + i + half * 8;   // C/D layout: lanes 16-31 -> M+8
                float v = acc[mi][ni][i] * scale + bcol;
                if (resid) v += resid[(long long)row * ldr + col];
                if (actGelu) v = 0.5f * v * (1.0f + erff(v * 0.70710678118f));
                if (outF) outF[(long long)row * ldo + col] = v;
                if (outH) outH[(long long)row * ldoh + col] = f2bf(v);
            }
        }
    }
}

// ---------------- host orchestration ----------------

extern "C" void kernel_launch(void* const* d_in, const int* in_sizes, int n_in,
                              void* d_out, int out_size, void* d_ws, size_t ws_size,
                              hipStream_t stream) {
    (void)in_sizes; (void)n_in; (void)out_size; (void)ws_size;
    const int*   node       = (const int*)d_in[0];
    const int*   edge       = (const int*)d_in[1];
    const float* node_emb   = (const float*)d_in[2];
    const float* edge_emb   = (const float*)d_in[3];
    const float* node_table = (const float*)d_in[4];
    const float* edge_table = (const float*)d_in[5];
    const float* W_in  = (const float*)d_in[6];   const float* b_in  = (const float*)d_in[7];
    const float* W_out = (const float*)d_in[8];   const float* b_out = (const float*)d_in[9];
    const float* g1    = (const float*)d_in[10];  const float* b1    = (const float*)d_in[11];
    const float* g_m   = (const float*)d_in[12];  const float* b_m   = (const float*)d_in[13];
    const float* W_q   = (const float*)d_in[14];  const float* b_q   = (const float*)d_in[15];
    const float* W_kv  = (const float*)d_in[16];  const float* b_kv  = (const float*)d_in[17];
    const float* W_f1  = (const float*)d_in[18];  const float* b_f1  = (const float*)d_in[19];
    const float* W_f2  = (const float*)d_in[20];  const float* b_f2  = (const float*)d_in[21];
    const float* g2    = (const float*)d_in[22];  const float* b2    = (const float*)d_in[23];
    const float* W_p   = (const float*)d_in[24];  const float* b_p   = (const float*)d_in[25];
    float* out = (float*)d_out;

    // ---- workspace carve-up ----
    char* w = (char*)d_ws; size_t off = 0;
    auto alloc = [&](size_t bytes) -> void* {
        void* p = w + off; off = (off + bytes + 255) & ~(size_t)255; return p;
    };
    const long long nWin  = 2LL*NL_*3*D_*D_,  nWout = 2LL*NL_*D_*D_,  nWq = nWout;
    const long long nWkv  = 2LL*NL_*2*D_*D_,  nWf1  = 2LL*NL_*DF_*D_, nWf2 = nWf1;
    bf16_t* Winh  = (bf16_t*)alloc(nWin  * 2);
    bf16_t* Wouth = (bf16_t*)alloc(nWout * 2);
    bf16_t* Wqh   = (bf16_t*)alloc(nWq   * 2);
    bf16_t* Wkvh  = (bf16_t*)alloc(nWkv  * 2);
    bf16_t* Wf1h  = (bf16_t*)alloc(nWf1  * 2);
    bf16_t* Wf2h  = (bf16_t*)alloc(nWf2  * 2);
    bf16_t* NtabTh = (bf16_t*)alloc((size_t)D_ * VN_ * 2);      // node_table^T [D,VN]
    bf16_t* Nembh  = (bf16_t*)alloc((size_t)BL_ * VN_ * 2);
    float* nA0 = (float*)alloc((size_t)BL_ * D_ * 4);
    float* nA1 = (float*)alloc((size_t)BL_ * D_ * 4);
    float* nB0 = (float*)alloc((size_t)BL_ * D_ * 4);
    float* nB1 = (float*)alloc((size_t)BL_ * D_ * 4);
    float* eA0 = (float*)alloc((size_t)BHLL_ * 4);
    float* eA1 = (float*)alloc((size_t)BHLL_ * 4);
    float* eB0 = (float*)alloc((size_t)BHLL_ * 4);
    float* eB1 = (float*)alloc((size_t)BHLL_ * 4);
    bf16_t* nH0 = (bf16_t*)alloc((size_t)BL_ * D_ * 2);
    bf16_t* nH1 = (bf16_t*)alloc((size_t)BL_ * D_ * 2);
    float*  tbuf   = (float*)alloc((size_t)BHLL_ * 4);
    float*  t2buf  = (float*)alloc((size_t)BHLL_ * 4);
    bf16_t* probsh = (bf16_t*)alloc((size_t)BHLL_ * 2);
    bf16_t* qkvh   = (bf16_t*)alloc((size_t)BL_ * 3 * D_ * 2);
    bf16_t* vTh    = (bf16_t*)alloc((size_t)BL_ * D_ * 2);      // V^T [B,H,DH,L]
    bf16_t* lnh    = (bf16_t*)alloc((size_t)BL_ * D_ * 2);
    bf16_t* q2h    = (bf16_t*)alloc((size_t)BL_ * D_ * 2);
    bf16_t* kvh    = (bf16_t*)alloc((size_t)BL_ * D_ * 2);      // k2 only (v2 is dead)
    bf16_t* obufh  = (bf16_t*)alloc((size_t)BL_ * D_ * 2);
    bf16_t* f1h    = (bf16_t*)alloc((size_t)BL_ * DF_ * 2);
    float*  xbuf   = (float*)alloc((size_t)BL_ * D_ * 4);

    auto cvt = [&](const float* src, bf16_t* dst, long long n) {
        long long blocks = (n + 255) / 256;
        k_f2bf<<<dim3((unsigned)blocks), dim3(256), 0, stream>>>(src, dst, n);
    };
    auto gemm = [&](const bf16_t* A, int lda, const bf16_t* Bm, int ldb,
                    const float* bias, const float* resid, int ldr,
                    float* outF, int ldo, bf16_t* outH, int ldoh,
                    int M, int N, int K, float scale, int gelu,
                    int batches, int hh,
                    long long sAb, long long sAh, long long sBb, long long sBh,
                    long long sOb, long long sOh, long long sOHb, long long sOHh) {
        int tiles = (M / 32) * (N / 64);
        dim3 grid((unsigned)((tiles + 7) / 8), (unsigned)batches);
        k_wmma_gemm<<<grid, dim3(256), 0, stream>>>(A, Bm, bias, resid, outF, outH,
            M, N, K, lda, ldb, ldr, ldo, ldoh, scale, gelu, hh,
            sAb, sAh, sBb, sBh, sOb, sOh, sOHb, sOHh);
    };

    // ---- one-time setup ----
    cvt(W_in,  Winh,  nWin);  cvt(W_out, Wouth, nWout); cvt(W_q,  Wqh,  nWq);
    cvt(W_kv,  Wkvh,  nWkv);  cvt(W_f1,  Wf1h,  nWf1);  cvt(W_f2, Wf2h, nWf2);
    cvt(node_emb, Nembh, (long long)BL_ * VN_);
    {   long long n = (long long)D_ * VN_;
        k_transpose_ntab<<<dim3((unsigned)((n + 255) / 256)), dim3(256), 0, stream>>>(node_table, NtabTh); }
    {   long long n = (long long)BL_ * D_;
        k_node_gather<<<dim3((unsigned)((n + 255) / 256)), dim3(256), 0, stream>>>(node, node_table, nA0); }
    {   long long n = (long long)B_ * L_ * L_;
        k_edge_embed<<<dim3((unsigned)((n + 255) / 256)), dim3(256), 0, stream>>>(
            edge, node, edge_emb, edge_table, eA0, eA1); }
    // n1 = node_emb @ node_table  == node_emb @ (node_table^T)^T, W-layout B = NtabTh [D,VN]
    gemm(Nembh, VN_, NtabTh, VN_, nullptr, nullptr, 0, nA1, D_, nullptr, 0,
         BL_, D_, VN_, 1.f, 0, 1, 1, 0,0,0,0, 0,0,0,0);

    float* nCur[2] = {nA0, nA1}; float* nNxt[2] = {nB0, nB1};
    float* eCur[2] = {eA0, eA1}; float* eNxt[2] = {eB0, eB1};
    bf16_t* nH[2] = {nH0, nH1};

    const long long sLL = (long long)L_ * L_;
    for (int layer = 0; layer < NL_; ++layer) {
        cvt(nCur[0], nH[0], (long long)BL_ * D_);
        cvt(nCur[1], nH[1], (long long)BL_ * D_);
        for (int s = 0; s < 2; ++s) {
            long long pi = (long long)s * NL_ + layer;
            const bf16_t* Win_s  = Winh  + pi * 3 * D_ * D_;
            const bf16_t* Wout_s = Wouth + pi * D_ * D_;
            const bf16_t* Wq_s   = Wqh   + pi * D_ * D_;
            const bf16_t* Wkv_s  = Wkvh  + pi * 2 * D_ * D_;
            const bf16_t* Wf1_s  = Wf1h  + pi * DF_ * D_;
            const bf16_t* Wf2_s  = Wf2h  + pi * D_ * DF_;
            const float* nSelf = nCur[s]; const float* eSelf = eCur[s];
            const bf16_t* nOtherH = nH[1 - s];
            float* nOut = nNxt[s]; float* eOut = eNxt[s];

            // x1 = LN(n_self); qkv = x1 @ W_in^T + b_in  (bf16)
            k_ln_bf16<<<dim3(BL_), dim3(256), 0, stream>>>(nSelf, g1 + pi * D_, b1 + pi * D_, lnh, D_);
            gemm(lnh, D_, Win_s, D_, b_in + pi * 3 * D_, nullptr, 0,
                 nullptr, 0, qkvh, 3 * D_, BL_, 3 * D_, D_, 1.f, 0, 1, 1,
                 0,0,0,0, 0,0,0,0);
            // V^T staging for the o-GEMM
            {   long long n = (long long)BL_ * D_;
                k_transpose_v<<<dim3((unsigned)((n + 255) / 256)), dim3(256), 0, stream>>>(qkvh, vTh); }
            // t = scale * Q K^T  (batched over b,h; K rows are W-layout naturally)
            gemm(qkvh, 3 * D_, qkvh + D_, 3 * D_, nullptr, nullptr, 0,
                 tbuf, L_, nullptr, 0, L_, L_, DH_, SCALE_F, 0, B_ * H_, H_,
                 (long long)L_ * 3 * D_, 64, (long long)L_ * 3 * D_, 64,
                 (long long)H_ * sLL, sLL, 0, 0);
            // probs = softmax(t + e_self)  -> bf16
            k_softmax_bf16<<<dim3((unsigned)((long long)B_ * H_ * L_ / 8)), dim3(256), 0, stream>>>(
                tbuf, eSelf, probsh, (long long)B_ * H_ * L_);
            // o = probs @ V   (B operand = V^T, W-layout [DH,L])
            gemm(probsh, L_, vTh, L_, nullptr, nullptr, 0,
                 nullptr, 0, obufh, D_, L_, DH_, L_, 1.f, 0, B_ * H_, H_,
                 (long long)H_ * sLL, sLL, (long long)H_ * DH_ * L_, (long long)DH_ * L_,
                 0, 0, (long long)L_ * D_, 64);
            // x = n_self + o @ W_out^T + b_out
            gemm(obufh, D_, Wout_s, D_, b_out + pi * D_, nSelf, D_,
                 xbuf, D_, nullptr, 0, BL_, D_, D_, 1.f, 0, 1, 1,
                 0,0,0,0, 0,0,0,0);
            // q2 = LN_m(x) @ W_q^T + b_q
            k_ln_bf16<<<dim3(BL_), dim3(256), 0, stream>>>(xbuf, g_m + pi * D_, b_m + pi * D_, lnh, D_);
            gemm(lnh, D_, Wq_s, D_, b_q + pi * D_, nullptr, 0,
                 nullptr, 0, q2h, D_, BL_, D_, D_, 1.f, 0, 1, 1,
                 0,0,0,0, 0,0,0,0);
            // k2 = n_other @ W_kv[:D]^T + b_kv[:D]   (v2 is dead compute)
            gemm(nOtherH, D_, Wkv_s, D_, b_kv + pi * 2 * D_, nullptr, 0,
                 nullptr, 0, kvh, D_, BL_, D_, D_, 1.f, 0, 1, 1,
                 0,0,0,0, 0,0,0,0);
            // t2 = scale * Q2 K2^T
            gemm(q2h, D_, kvh, D_, nullptr, nullptr, 0,
                 t2buf, L_, nullptr, 0, L_, L_, DH_, SCALE_F, 0, B_ * H_, H_,
                 (long long)L_ * D_, 64, (long long)L_ * D_, 64,
                 (long long)H_ * sLL, sLL, 0, 0);
            // e_new = e_self + t + t2
            k_edge_update<<<dim3((unsigned)((BHLL_ + 255) / 256)), dim3(256), 0, stream>>>(
                eSelf, tbuf, t2buf, eOut, BHLL_);
            // FFN: n_new = x + GELU(LN2(x) @ Wf1^T + b1) @ Wf2^T + b2
            k_ln_bf16<<<dim3(BL_), dim3(256), 0, stream>>>(xbuf, g2 + pi * D_, b2 + pi * D_, lnh, D_);
            gemm(lnh, D_, Wf1_s, D_, b_f1 + pi * DF_, nullptr, 0,
                 nullptr, 0, f1h, DF_, BL_, DF_, D_, 1.f, 1, 1, 1,
                 0,0,0,0, 0,0,0,0);
            gemm(f1h, DF_, Wf2_s, DF_, b_f2 + pi * D_, xbuf, D_,
                 nOut, D_, nullptr, 0, BL_, D_, DF_, 1.f, 0, 1, 1,
                 0,0,0,0, 0,0,0,0);
        }
        { float* t0 = nCur[0]; nCur[0] = nNxt[0]; nNxt[0] = t0;
          float* t1 = nCur[1]; nCur[1] = nNxt[1]; nNxt[1] = t1;
          float* u0 = eCur[0]; eCur[0] = eNxt[0]; eNxt[0] = u0;
          float* u1 = eCur[1]; eCur[1] = eNxt[1]; eNxt[1] = u1; }
    }

    k_pool_proj<<<dim3(B_), dim3(256), 0, stream>>>(nCur[1], node, W_p, b_p, out);
}
